// SelfAttention_66949950210532
// MI455X (gfx1250) — matrix operand order repaired and checked by reading, not compiled, not run on hardware
//
#include <hip/hip_runtime.h>
#include <hip/hip_bf16.h>

typedef __attribute__((ext_vector_type(16))) __bf16 v16bf;
typedef __attribute__((ext_vector_type(8)))  __bf16 v8bf;
typedef __attribute__((ext_vector_type(8)))  float  v8f;

typedef __attribute__((address_space(3))) void lds_void_t;
typedef __attribute__((address_space(1))) void glb_void_t;

// builtin's b128 payload type: int __vector(4) with matching address spaces
typedef int b128_t __attribute__((vector_size(16)));
typedef __attribute__((address_space(1))) b128_t glb_b128_t;
typedef __attribute__((address_space(3))) b128_t lds_b128_t;

#if __has_builtin(__builtin_amdgcn_global_load_async_to_lds_b128)
#define USE_ASYNC_LDS 1
#else
#define USE_ASYNC_LDS 0
#endif

namespace {
constexpr int  kB  = 8;
constexpr int  kS  = 2048;
constexpr int  kD  = 1024;
constexpr long kBS = (long)kB * kS;   // 16384
}

__device__ static inline lds_void_t* as_lds(void* p) {
  return (lds_void_t*)p;
}
__device__ static inline glb_void_t* as_glb(const void* p) {
  return (glb_void_t*)const_cast<void*>(p);
}

#if USE_ASYNC_LDS
__device__ static inline void async_load_b128(const void* g, void* l) {
  __builtin_amdgcn_global_load_async_to_lds_b128(
      (glb_b128_t*)as_glb(g), (lds_b128_t*)as_lds(l), 0, 0);
}
#endif

__device__ static inline void wait_async_zero() {
#if __has_builtin(__builtin_amdgcn_s_wait_asynccnt)
  __builtin_amdgcn_s_wait_asynccnt(0);
#elif defined(__AMDGCN__)
  asm volatile("s_wait_asynccnt 0x0" ::: "memory");
#endif
}

// ---------------------------------------------------------------------------
// f32 -> bf16 conversion, 4 elements per thread (float4 in, 8B out)
// ---------------------------------------------------------------------------
__global__ __launch_bounds__(256)
void cvt_f32_to_bf16(const float* __restrict__ src, __bf16* __restrict__ dst,
                     int n4) {
  int i = blockIdx.x * blockDim.x + threadIdx.x;
  if (i >= n4) return;
  float4 f = reinterpret_cast<const float4*>(src)[i];
  union { __bf16 h[4]; uint2 u; } p;
  p.h[0] = (__bf16)f.x; p.h[1] = (__bf16)f.y;
  p.h[2] = (__bf16)f.z; p.h[3] = (__bf16)f.w;
  reinterpret_cast<uint2*>(dst)[i] = p.u;
}

// ---------------------------------------------------------------------------
// WMMA GEMM: C[m,n] = alpha * sum_k A[m,k] * B[n,k]   (A,B bf16, K-contiguous)
// MODE 0: f32 row-major store    MODE 1: bf16 row-major store
// MODE 2: bf16 transposed store  C[n,m]  (contiguous b128 per lane)
// Workgroup: 256 thr = 8 wave32 (4x2), C tile 128x128, 32x64 per wave,
// K step 32, double-buffered LDS fed by GLOBAL_LOAD_ASYNC_TO_LDS_B128.
// All problem dims here are multiples of the tile sizes -> no bounds checks.
// ---------------------------------------------------------------------------
template <int MODE>
__global__ __launch_bounds__(256)
void gemm_bf16_nt(const __bf16* __restrict__ A, int lda,
                  const __bf16* __restrict__ B, int ldb,
                  void* __restrict__ Cv, int ldc, int K, float alpha) {
  constexpr int MT = 128, NT = 128, KT = 32, LS = 40;  // LS: padded LDS row
  __shared__ __align__(16) __bf16 sA[2][MT * LS];
  __shared__ __align__(16) __bf16 sB[2][NT * LS];

  const int tid  = threadIdx.x;
  const int wave = tid >> 5, lane = tid & 31;
  const int wm = wave >> 1, wn = wave & 1;   // 4 x 2 wave grid
  const int ln = lane & 15, hl = lane >> 4;  // lane-in-16, half select

  const long m0 = (long)blockIdx.x * MT;
  const long n0 = (long)blockIdx.y * NT;

  // cooperative staging: 128 rows x 32 cols per tile, 2 x b128 per thread
  const int srow = tid >> 1, scol = (tid & 1) * 16;
  const __bf16* gA = A + (m0 + srow) * (long)lda + scol;
  const __bf16* gB = B + (n0 + srow) * (long)ldb + scol;
  __bf16* lA = &sA[0][srow * LS + scol];
  __bf16* lB = &sB[0][srow * LS + scol];

  v8f acc[2][4] = {};
  const int iters = K / KT;

  auto stage = [&](int kt, int buf) {
    const __bf16* ga = gA + kt * KT;
    const __bf16* gb = gB + kt * KT;
    __bf16* la = lA + buf * (MT * LS);
    __bf16* lb = lB + buf * (NT * LS);
#if USE_ASYNC_LDS
    async_load_b128(ga,     la);
    async_load_b128(ga + 8, la + 8);
    async_load_b128(gb,     lb);
    async_load_b128(gb + 8, lb + 8);
#else
    v8bf a0 = *(const v8bf*)ga, a1 = *(const v8bf*)(ga + 8);
    v8bf b0 = *(const v8bf*)gb, b1 = *(const v8bf*)(gb + 8);
    *(v8bf*)la = a0;       *(v8bf*)(la + 8) = a1;
    *(v8bf*)lb = b0;       *(v8bf*)(lb + 8) = b1;
#endif
  };

  stage(0, 0);
#if USE_ASYNC_LDS
  wait_async_zero();
#endif
  __syncthreads();

  for (int kt = 0; kt < iters; ++kt) {
    const int cur = kt & 1;
    if (kt + 1 < iters) stage(kt + 1, cur ^ 1);
    if (kt + 2 < iters) {  // deep prefetch into L2 (global_prefetch_b8)
      __builtin_prefetch(gA + (kt + 2) * KT, 0, 1);
      __builtin_prefetch(gB + (kt + 2) * KT, 0, 1);
    }

    // A fragment: lane -> row m=l&15, K chunks at [hl*8,+8) and [16+hl*8,+8)
    v16bf af[2], bfm[4];
#pragma unroll
    for (int i = 0; i < 2; ++i) {
      const __bf16* p = &sA[cur][(wm * 32 + i * 16 + ln) * LS];
      v8bf lo = *(const v8bf*)(p + hl * 8);
      v8bf hi = *(const v8bf*)(p + 16 + hl * 8);
      af[i] = __builtin_shufflevector(lo, hi, 0, 1, 2, 3, 4, 5, 6, 7, 8, 9, 10,
                                      11, 12, 13, 14, 15);
    }
    // B fragment: lane -> col n=l&15, K chunks at [hl*8,+8) and [16+hl*8,+8)
#pragma unroll
    for (int j = 0; j < 4; ++j) {
      const __bf16* p = &sB[cur][(wn * 64 + j * 16 + ln) * LS];
      v8bf lo = *(const v8bf*)(p + hl * 8);
      v8bf hi = *(const v8bf*)(p + 16 + hl * 8);
      bfm[j] = __builtin_shufflevector(lo, hi, 0, 1, 2, 3, 4, 5, 6, 7, 8, 9, 10,
                                       11, 12, 13, 14, 15);
    }

#pragma unroll
    for (int i = 0; i < 2; ++i)
#pragma unroll
      for (int j = 0; j < 4; ++j)
        acc[i][j] = __builtin_amdgcn_wmma_f32_16x16x32_bf16(
            false, af[i], false, bfm[j], (short)0, acc[i][j], false, false);

#if USE_ASYNC_LDS
    if (kt + 1 < iters) wait_async_zero();
#endif
    __syncthreads();
  }

  // Epilogue. C VGPR layout: lane -> n=l&15, VGPR g -> m = g + 8*hl.
  const long cm = m0 + wm * 32, cn = n0 + wn * 64;
  if constexpr (MODE == 0) {
    float* C = (float*)Cv;
#pragma unroll
    for (int i = 0; i < 2; ++i)
#pragma unroll
      for (int j = 0; j < 4; ++j)
#pragma unroll
        for (int g = 0; g < 8; ++g)
          C[(cm + i * 16 + g + 8 * hl) * (long)ldc + cn + j * 16 + ln] =
              alpha * acc[i][j][g];
  } else if constexpr (MODE == 1) {
    __bf16* C = (__bf16*)Cv;
#pragma unroll
    for (int i = 0; i < 2; ++i)
#pragma unroll
      for (int j = 0; j < 4; ++j)
#pragma unroll
        for (int g = 0; g < 8; ++g)
          C[(cm + i * 16 + g + 8 * hl) * (long)ldc + cn + j * 16 + ln] =
              (__bf16)(alpha * acc[i][j][g]);
  } else {  // MODE 2: transposed store -> per lane 8 contiguous bf16 = b128
    __bf16* C = (__bf16*)Cv;
#pragma unroll
    for (int i = 0; i < 2; ++i)
#pragma unroll
      for (int j = 0; j < 4; ++j) {
        union { __bf16 h[8]; uint4 u; } pk;
#pragma unroll
        for (int g = 0; g < 8; ++g) pk.h[g] = (__bf16)(alpha * acc[i][j][g]);
        const long row = cn + j * 16 + ln;       // transposed row = n
        const long col = cm + i * 16 + 8 * hl;   // 16B-aligned column base
        *reinterpret_cast<uint4*>(&C[row * (long)ldc + col]) = pk.u;
      }
  }
}

// ---------------------------------------------------------------------------
// Row softmax over 2048 f32 scores -> bf16 attn. One block (8 wave32) per row.
// ---------------------------------------------------------------------------
__global__ __launch_bounds__(256)
void softmax_row_bf16(const float* __restrict__ Sc, __bf16* __restrict__ At,
                      float scale) {
  const long row = blockIdx.x;
  const float* sp = Sc + row * kS;
  const int t = threadIdx.x;
  const int wave = t >> 5, lane = t & 31;

  float v[8];
  float mx = -3.0e38f;
#pragma unroll
  for (int i = 0; i < 8; ++i) {
    v[i] = sp[t + i * 256] * scale;
    mx = fmaxf(mx, v[i]);
  }
#pragma unroll
  for (int off = 16; off > 0; off >>= 1) mx = fmaxf(mx, __shfl_xor(mx, off, 32));

  __shared__ float rmax[8], rsum[8];
  if (lane == 0) rmax[wave] = mx;
  __syncthreads();
  mx = rmax[0];
#pragma unroll
  for (int w = 1; w < 8; ++w) mx = fmaxf(mx, rmax[w]);

  float s = 0.f;
#pragma unroll
  for (int i = 0; i < 8; ++i) {
    v[i] = __expf(v[i] - mx);
    s += v[i];
  }
#pragma unroll
  for (int off = 16; off > 0; off >>= 1) s += __shfl_xor(s, off, 32);
  if (lane == 0) rsum[wave] = s;
  __syncthreads();
  s = 0.f;
#pragma unroll
  for (int w = 0; w < 8; ++w) s += rsum[w];
  const float inv = 1.0f / s;

  __bf16* op = At + row * kS;
#pragma unroll
  for (int i = 0; i < 8; ++i) op[t + i * 256] = (__bf16)(v[i] * inv);
}

// ---------------------------------------------------------------------------
extern "C" void kernel_launch(void* const* d_in, const int* in_sizes, int n_in,
                              void* d_out, int out_size, void* d_ws,
                              size_t ws_size, hipStream_t stream) {
  (void)in_sizes; (void)n_in; (void)out_size; (void)ws_size;
  const float* x  = (const float*)d_in[0];
  const float* Wq = (const float*)d_in[1];
  const float* Wk = (const float*)d_in[2];
  const float* Wv = (const float*)d_in[3];
  const float* Wo = (const float*)d_in[4];

  char* ws = (char*)d_ws;
  size_t off = 0;
  auto take = [&](size_t bytes) -> void* {
    void* p = ws + off;
    off += (bytes + 255) & ~(size_t)255;
    return p;
  };
  __bf16* xh  = (__bf16*)take((size_t)kBS * kD * 2);
  __bf16* Wqh = (__bf16*)take((size_t)kD * kD * 2);
  __bf16* Wkh = (__bf16*)take((size_t)kD * kD * 2);
  __bf16* Wvh = (__bf16*)take((size_t)kD * kD * 2);
  __bf16* Woh = (__bf16*)take((size_t)kD * kD * 2);
  __bf16* Qh  = (__bf16*)take((size_t)kBS * kD * 2);
  __bf16* Kh  = (__bf16*)take((size_t)kBS * kD * 2);
  __bf16* Vt  = (__bf16*)take((size_t)kBS * kD * 2);  // [kD][kBS], ld = kBS
  float*  Sc  = (float*)take((size_t)kS * kS * 4);    // per-batch scores (f32)
  __bf16* At  = (__bf16*)take((size_t)kS * kS * 2);   // per-batch attn (bf16)
  __bf16* Wt  = xh;  // reuse xh region for attn-weighted values (stream-ordered)

  // 1) convert inputs to bf16
  int n4 = (int)(kBS * kD / 4);
  cvt_f32_to_bf16<<<(n4 + 255) / 256, 256, 0, stream>>>(x, xh, n4);
  n4 = kD * kD / 4;
  cvt_f32_to_bf16<<<(n4 + 255) / 256, 256, 0, stream>>>(Wq, Wqh, n4);
  cvt_f32_to_bf16<<<(n4 + 255) / 256, 256, 0, stream>>>(Wk, Wkh, n4);
  cvt_f32_to_bf16<<<(n4 + 255) / 256, 256, 0, stream>>>(Wv, Wvh, n4);
  cvt_f32_to_bf16<<<(n4 + 255) / 256, 256, 0, stream>>>(Wo, Woh, n4);

  // 2) Q/K/V projections (y = x @ W^T).  V stored transposed: Vt[e, b*S+s].
  dim3 gProj((unsigned)(kBS / 128), (unsigned)(kD / 128));
  gemm_bf16_nt<1><<<gProj, 256, 0, stream>>>(xh, kD, Wqh, kD, Qh, kD, kD, 1.0f);
  gemm_bf16_nt<1><<<gProj, 256, 0, stream>>>(xh, kD, Wkh, kD, Kh, kD, kD, 1.0f);
  gemm_bf16_nt<2><<<gProj, 256, 0, stream>>>(xh, kD, Wvh, kD, Vt, (int)kBS, kD,
                                             1.0f);

  // 3) per-batch attention: scores -> softmax -> attn @ V
  const float scale = 0.03125f;  // 1/sqrt(1024)
  for (int b = 0; b < kB; ++b) {
    const __bf16* Qb = Qh + (size_t)b * kS * kD;
    const __bf16* Kb = Kh + (size_t)b * kS * kD;
    const __bf16* Vb = Vt + (size_t)b * kS;         // column slice of [kD][kBS]
    __bf16*       Wb = Wt + (size_t)b * kS * kD;
    gemm_bf16_nt<0><<<dim3(kS / 128, kS / 128), 256, 0, stream>>>(
        Qb, kD, Kb, kD, Sc, kS, kD, 1.0f);
    softmax_row_bf16<<<kS, 256, 0, stream>>>(Sc, At, scale);
    gemm_bf16_nt<1><<<dim3(kS / 128, kD / 128), 256, 0, stream>>>(
        At, kS, Vb, (int)kBS, Wb, kD, kS, 1.0f);
  }

  // 4) output projection -> f32 d_out
  gemm_bf16_nt<0><<<gProj, 256, 0, stream>>>(Wt, kD, Woh, kD, (float*)d_out, kD,
                                             kD, 1.0f);
}
